// MoRAttention_55250459296378
// MI455X (gfx1250) — matrix operand order repaired and checked by reading, not compiled, not run on hardware
//
#include <hip/hip_runtime.h>
#include <cstdint>

typedef __bf16 bf16_t;
typedef __attribute__((ext_vector_type(16))) __bf16 v16bf;
typedef __attribute__((ext_vector_type(8)))  float  v8f;

#define B_   2
#define S_   1024
#define D_   2048
#define H_   16
#define HKV_ 4
#define HD_  128
#define QD_  (H_*HD_)   // 2048
#define KD_  (HKV_*HD_) // 512

// ---------------------------------------------------------------------------
// CDNA5 async copy: global -> LDS, 16B per lane, tracked by ASYNCcnt.
// (CDNA5 ISA; VGLOBAL opcode 98)
// ---------------------------------------------------------------------------
__device__ __forceinline__ void async_copy_b128(const void* g, void* lds) {
  uint32_t l = (uint32_t)(size_t)lds;          // low 32 bits = LDS offset
  asm volatile("global_load_async_to_lds_b128 %0, %1, off"
               :: "v"(l), "v"(g) : "memory");
}
__device__ __forceinline__ void wait_async_le1() {
  asm volatile("s_wait_asynccnt 0x1" ::: "memory");
}
__device__ __forceinline__ void wait_async_0() {
  asm volatile("s_wait_asynccnt 0x0" ::: "memory");
}

// ---------------------------------------------------------------------------
// WMMA fragment loaders (wave32; contiguous k = 16*half + i order, used by A and B alike)
// A 16x32 bf16: lanes 0-15 -> row M=lane, K=0..15;
//               lanes 16-31 -> row M=lane-16, K=16..31.
// ---------------------------------------------------------------------------
__device__ __forceinline__ v16bf load_a_frag(const bf16_t* __restrict__ A,
                                             long lda, int row0, int k0) {
  int lane = threadIdx.x & 31;
  int half = lane >> 4;
  int r    = row0 + (lane & 15);
  const bf16_t* p = A + (long)r * lda + k0 + half * 16;
  v16bf f;
#pragma unroll
  for (int i = 0; i < 16; ++i) f[i] = p[i];
  return f;
}

// B 32x16 from K-contiguous storage: element (k,n) at Bt[n*ldk + k].
// Lane = column n; half 0 -> K=0..15, half 1 -> K=16..31 (16 contiguous bf16).
__device__ __forceinline__ v16bf load_bt_frag(const bf16_t* __restrict__ Bt,
                                              long ldk, int k0, int n0) {
  int lane = threadIdx.x & 31;
  int half = lane >> 4;
  const bf16_t* p = Bt + (long)(n0 + (lane & 15)) * ldk + k0 + half * 16;
  v16bf f;
#pragma unroll
  for (int i = 0; i < 16; ++i) f[i] = p[i];
  return f;
}

// Same fragment shape read from an LDS-staged 64x32 slab (n-major, ldk=32).
__device__ __forceinline__ v16bf load_bt_frag_lds(const bf16_t* base, int n0) {
  int lane = threadIdx.x & 31;
  int half = lane >> 4;
  const bf16_t* p = base + (n0 + (lane & 15)) * 32 + half * 16;
  v16bf f;
#pragma unroll
  for (int i = 0; i < 16; ++i) f[i] = p[i];
  return f;
}

// ---------------------------------------------------------------------------
// fp32 -> bf16 conversion (activations)
// ---------------------------------------------------------------------------
__global__ __launch_bounds__(256) void cvt_bf16_kernel(const float* __restrict__ x,
                                                       bf16_t* __restrict__ y, int n) {
  int i = blockIdx.x * blockDim.x + threadIdx.x;
  if (i < n) y[i] = (bf16_t)x[i];
}

// ---------------------------------------------------------------------------
// fp32 W[K][N] -> bf16 Wt[N][K], 32x32 LDS tile transpose
// block = (32,8), grid = (K/32, N/32)
// ---------------------------------------------------------------------------
__global__ __launch_bounds__(256) void cvt_transpose_kernel(const float* __restrict__ W,
                                                            bf16_t* __restrict__ Wt,
                                                            int K, int N) {
  __shared__ bf16_t tile[32][33];
  int tk = blockIdx.x * 32;
  int tn = blockIdx.y * 32;
  int tx = threadIdx.x, ty = threadIdx.y;
#pragma unroll
  for (int r = ty; r < 32; r += 8)
    tile[r][tx] = (bf16_t)W[(long)(tk + r) * N + tn + tx];
  __syncthreads();
#pragma unroll
  for (int r = ty; r < 32; r += 8)
    Wt[(long)(tn + r) * K + tk + tx] = tile[tx][r];
}

// ---------------------------------------------------------------------------
// bf16 GEMM, NT form: C(MxN) = A(MxK) * Bt(NxK)^T.
// Block = 8 waves = 128x64 tile over a SHARED 64-col B strip. B k-slices
// (64x32 bf16 = 4KB) are DMA'd to LDS with global_load_async_to_lds_b128,
// double-buffered. All 4 B fragments are loaded as one ds clause BEFORE the
// WMMA chain so waits are partial instead of full dscnt-0 stalls.
// ---------------------------------------------------------------------------
__global__ __launch_bounds__(256) void gemm_nt_kernel(const bf16_t* __restrict__ A,
                                                      const bf16_t* __restrict__ Bt,
                                                      float* __restrict__ C,
                                                      int M, int N, int K) {
  __shared__ bf16_t ldsB[2][64 * 32];
  int t = threadIdx.x;
  int w = t >> 5;
  int nblk = N >> 6;
  int bm = (blockIdx.x / nblk) * 128;
  int bn = (blockIdx.x % nblk) * 64;
  int tm = bm + w * 16;

  // staging: thread t moves 8 bf16 (16B): row n = t>>2, k-offset = (t&3)*8
  int sn = t >> 2, sk = (t & 3) * 8;
  const bf16_t* gB = Bt + (long)(bn + sn) * K + sk;
  bf16_t* lB0 = &ldsB[0][sn * 32 + sk];
  bf16_t* lB1 = &ldsB[1][sn * 32 + sk];

  async_copy_b128(gB, lB0);                     // stage k = 0

  v8f acc[4];
#pragma unroll
  for (int j = 0; j < 4; ++j) acc[j] = (v8f){};

  for (int k = 0; k < K; k += 32) {
    int buf = (k >> 5) & 1;
    if (k + 32 < K) {
      async_copy_b128(gB + (k + 32), buf ? lB0 : lB1);   // stage next
      __builtin_prefetch(gB + (k + 96), 0, 1);           // L2 prefetch ahead
      wait_async_le1();                                  // current slab done
    } else {
      wait_async_0();
    }
    __syncthreads();                                     // slab visible to all

    v16bf a = load_a_frag(A, K, tm, k);
    v16bf bfr[4];
#pragma unroll
    for (int j = 0; j < 4; ++j)                          // one ds clause
      bfr[j] = load_bt_frag_lds(&ldsB[buf][0], j * 16);
#pragma unroll
    for (int j = 0; j < 4; ++j)
      acc[j] = __builtin_amdgcn_wmma_f32_16x16x32_bf16(false, a, false, bfr[j],
                                                       (short)0, acc[j], false, false);
    __syncthreads();                                     // done reading slab
  }

  int lane = threadIdx.x & 31;
  int half = lane >> 4, ln = lane & 15;
#pragma unroll
  for (int j = 0; j < 4; ++j)
#pragma unroll
    for (int r = 0; r < 8; ++r)
      C[(long)(tm + 8 * half + r) * N + bn + j * 16 + ln] = acc[j][r];
}

// ---------------------------------------------------------------------------
// Per-batch stable partition (active tokens first) via LDS Hillis-Steele scan.
// ---------------------------------------------------------------------------
__global__ __launch_bounds__(1024) void build_order_kernel(const unsigned char* __restrict__ mask,
                                                           const int* __restrict__ pos_ids,
                                                           int* __restrict__ order,
                                                           int* __restrict__ pos_sel) {
  __shared__ int sm[S_];
  int b = blockIdx.x, t = threadIdx.x;
  int a = mask[b * S_ + t] ? 1 : 0;
  sm[t] = a;
  __syncthreads();
  for (int off = 1; off < S_; off <<= 1) {
    int v = sm[t];
    int u = (t >= off) ? sm[t - off] : 0;
    __syncthreads();
    sm[t] = v + u;
    __syncthreads();
  }
  int incl = sm[t];
  int na   = sm[S_ - 1];
  int excl = incl - a;
  int rank = a ? excl : (na + (t - excl));
  order[b * S_ + rank]   = t;
  pos_sel[b * S_ + rank] = a ? pos_ids[b * S_ + t] : -1;
}

// ---------------------------------------------------------------------------
// RoPE on Q (fp32 -> bf16). One thread per (b,s,h,d<64).
// ---------------------------------------------------------------------------
__global__ __launch_bounds__(256) void rope_q_kernel(const float* __restrict__ Qf,
                                                     const float* __restrict__ cosT,
                                                     const float* __restrict__ sinT,
                                                     bf16_t* __restrict__ Qbf) {
  long i = (long)blockIdx.x * blockDim.x + threadIdx.x;   // B*S*H*64
  int  d = (int)(i & 63);
  long rest = i >> 6;
  int  h = (int)(rest & (H_ - 1));
  long bs = rest >> 4;
  int  s = (int)(bs & (S_ - 1));
  const float* q = Qf + bs * QD_ + h * HD_;
  float c0 = cosT[s * HD_ + d],      sn0 = sinT[s * HD_ + d];
  float c1 = cosT[s * HD_ + 64 + d], sn1 = sinT[s * HD_ + 64 + d];
  float x1 = q[d], x2 = q[d + 64];
  long o = bs * QD_ + h * HD_;
  Qbf[o + d]      = (bf16_t)(x1 * c0 - x2 * sn0);
  Qbf[o + 64 + d] = (bf16_t)(x2 * c1 + x1 * sn1);
}

// ---------------------------------------------------------------------------
// Gather K/V into compacted order; RoPE K at source position; zero inactive.
// K row-major [b][j][hkv][d]; V transposed Vt[b][hkv][d][j].
// ---------------------------------------------------------------------------
__global__ __launch_bounds__(256) void gather_kv_kernel(const float* __restrict__ Kf,
                                                        const float* __restrict__ Vf,
                                                        const int* __restrict__ order,
                                                        const int* __restrict__ pos_sel,
                                                        const float* __restrict__ cosT,
                                                        const float* __restrict__ sinT,
                                                        bf16_t* __restrict__ Ksel,
                                                        bf16_t* __restrict__ Vt) {
  long i = (long)blockIdx.x * blockDim.x + threadIdx.x;   // B*S*HKV*64
  int  d = (int)(i & 63);
  long rest = i >> 6;
  int  hk = (int)(rest & (HKV_ - 1));
  long bj = rest >> 2;
  int  j = (int)(bj & (S_ - 1));
  int  b = (int)(bj >> 10);
  int  src = order[b * S_ + j];
  bool ok  = pos_sel[b * S_ + j] >= 0;
  const float* k = Kf + ((long)b * S_ + src) * KD_ + hk * HD_;
  const float* v = Vf + ((long)b * S_ + src) * KD_ + hk * HD_;
  float c0 = cosT[src * HD_ + d],      sn0 = sinT[src * HD_ + d];
  float c1 = cosT[src * HD_ + 64 + d], sn1 = sinT[src * HD_ + 64 + d];
  float k1 = k[d], k2 = k[d + 64];
  long ko = ((long)b * S_ + j) * KD_ + hk * HD_;
  Ksel[ko + d]      = (bf16_t)(ok ? (k1 * c0 - k2 * sn0) : 0.f);
  Ksel[ko + 64 + d] = (bf16_t)(ok ? (k2 * c1 + k1 * sn1) : 0.f);
  long vbase = ((long)(b * HKV_ + hk) * HD_) * S_;        // Vt[b][hk][d][j]
  Vt[vbase + (long)d * S_ + j]        = (bf16_t)(ok ? v[d] : 0.f);
  Vt[vbase + (long)(d + 64) * S_ + j] = (bf16_t)(ok ? v[d + 64] : 0.f);
}

// ---------------------------------------------------------------------------
// Flash-style causal GQA attention. One wave per (b, h, 16-row Q tile).
// K fragments are loaded as one clause before the score WMMA chain; V
// fragments load before the softmax VALU work (overlapping exp/shuffles);
// next chunk is L2-prefetched. P round-trips through LDS (C->A layout).
// ---------------------------------------------------------------------------
#define ATT_WAVES 4
__global__ __launch_bounds__(128) void attn_kernel(const bf16_t* __restrict__ Q,
                                                   const bf16_t* __restrict__ K,
                                                   const bf16_t* __restrict__ Vt,
                                                   const int* __restrict__ pos_sel,
                                                   float scale,
                                                   bf16_t* __restrict__ AO) {
  __shared__ bf16_t pbuf[ATT_WAVES][16][32];
  int w    = threadIdx.x >> 5;
  int lane = threadIdx.x & 31;
  int half = lane >> 4, ln = lane & 15;

  int gw = blockIdx.x * ATT_WAVES + w;     // (b*H + h)*(S/16) + qt
  int qt = gw & 63;
  int h  = (gw >> 6) & (H_ - 1);
  int b  = gw >> 10;
  int hkv = h >> 2;

  const bf16_t* Qb  = Q  + ((long)b * S_ + qt * 16) * QD_ + h * HD_;
  const bf16_t* Kb  = K  + (long)b * S_ * KD_ + hkv * HD_;
  const bf16_t* Vtb = Vt + (long)(b * HKV_ + hkv) * HD_ * S_;
  const int*    ps  = pos_sel + b * S_;

  v16bf qf[4];
#pragma unroll
  for (int dc = 0; dc < 4; ++dc) qf[dc] = load_a_frag(Qb, QD_, 0, dc * 32);

  v8f o[8];
#pragma unroll
  for (int dt = 0; dt < 8; ++dt) o[dt] = (v8f){};
  float mrow[8], lrow[8];
#pragma unroll
  for (int r = 0; r < 8; ++r) { mrow[r] = -1e30f; lrow[r] = 0.f; }

  for (int kc = 0; kc < S_; kc += 32) {
    // ---- K fragments for both 16-key tiles, issued as one load clause ----
    v16bf kf0[4], kf1[4];
#pragma unroll
    for (int dc = 0; dc < 4; ++dc) {
      kf0[dc] = load_bt_frag(Kb, KD_, dc * 32, kc);
      kf1[dc] = load_bt_frag(Kb, KD_, dc * 32, kc + 16);
    }
    // ---- scores: contraction over HD in 4 chunks of 32 ----
    v8f s0 = {}, s1 = {};
#pragma unroll
    for (int dc = 0; dc < 4; ++dc) {
      s0 = __builtin_amdgcn_wmma_f32_16x16x32_bf16(false, qf[dc], false, kf0[dc],
                                                   (short)0, s0, false, false);
      s1 = __builtin_amdgcn_wmma_f32_16x16x32_bf16(false, qf[dc], false, kf1[dc],
                                                   (short)0, s1, false, false);
    }

    // ---- V fragments for this chunk: independent of P, overlap softmax ----
    v16bf vf[8];
#pragma unroll
    for (int dt = 0; dt < 8; ++dt) vf[dt] = load_bt_frag(Vtb, S_, kc, dt * 16);

    if (kc + 32 < S_) {
      __builtin_prefetch(Kb + (long)(kc + 32 + ln) * KD_, 0, 1);
      __builtin_prefetch(Vtb + (long)ln * S_ + kc + 32, 0, 1);
    }

    int ps0 = ps[kc + ln];
    int ps1 = ps[kc + 16 + ln];

    float p0[8], p1[8];
#pragma unroll
    for (int r = 0; r < 8; ++r) {
      int qpos = qt * 16 + 8 * half + r;
      bool ok0 = (ps0 >= 0) && (ps0 <= qpos);
      bool ok1 = (ps1 >= 0) && (ps1 <= qpos);
      float x0 = ok0 ? s0[r] * scale : -1e30f;
      float x1 = ok1 ? s1[r] * scale : -1e30f;
      float mx = fmaxf(x0, x1);
#pragma unroll
      for (int m = 1; m <= 8; m <<= 1) mx = fmaxf(mx, __shfl_xor(mx, m, 32));
      float mn = fmaxf(mrow[r], mx);
      float alpha = __expf(mrow[r] - mn);
      mrow[r] = mn;
      p0[r] = ok0 ? __expf(x0 - mn) : 0.f;
      p1[r] = ok1 ? __expf(x1 - mn) : 0.f;
      float rs = p0[r] + p1[r];
#pragma unroll
      for (int m = 1; m <= 8; m <<= 1) rs += __shfl_xor(rs, m, 32);
      lrow[r] = lrow[r] * alpha + rs;
#pragma unroll
      for (int dt = 0; dt < 8; ++dt) o[dt][r] *= alpha;
    }

    // ---- C-layout -> A-layout via LDS ----
    __syncthreads();
#pragma unroll
    for (int r = 0; r < 8; ++r) {
      pbuf[w][8 * half + r][ln]      = (bf16_t)p0[r];
      pbuf[w][8 * half + r][16 + ln] = (bf16_t)p1[r];
    }
    __syncthreads();
    v16bf pa;
    {
      const bf16_t* pr = &pbuf[w][ln][half * 16];
#pragma unroll
      for (int i = 0; i < 16; ++i) pa[i] = pr[i];
    }

    // ---- O += P * V ----
#pragma unroll
    for (int dt = 0; dt < 8; ++dt)
      o[dt] = __builtin_amdgcn_wmma_f32_16x16x32_bf16(false, pa, false, vf[dt],
                                                      (short)0, o[dt], false, false);
  }

  // ---- normalize (reciprocal) + store ----
#pragma unroll
  for (int r = 0; r < 8; ++r) {
    float linv = 1.0f / lrow[r];
    long row = (long)b * S_ + qt * 16 + 8 * half + r;
#pragma unroll
    for (int dt = 0; dt < 8; ++dt)
      AO[row * QD_ + h * HD_ + dt * 16 + ln] = (bf16_t)(o[dt][r] * linv);
  }
}

// ---------------------------------------------------------------------------
// Host-side launcher
// ---------------------------------------------------------------------------
extern "C" void kernel_launch(void* const* d_in, const int* in_sizes, int n_in,
                              void* d_out, int out_size, void* d_ws, size_t ws_size,
                              hipStream_t stream) {
  (void)in_sizes; (void)n_in; (void)out_size; (void)ws_size;

  const float*         hidden  = (const float*)d_in[0];
  const float*         cosT    = (const float*)d_in[1];
  const float*         sinT    = (const float*)d_in[2];
  const float*         wq      = (const float*)d_in[3];
  const float*         wk      = (const float*)d_in[4];
  const float*         wv      = (const float*)d_in[5];
  const float*         wo      = (const float*)d_in[6];
  const int*           pos_ids = (const int*)d_in[7];
  const unsigned char* amask   = (const unsigned char*)d_in[8];
  float* out = (float*)d_out;

  char* p = (char*)d_ws;
  auto alloc = [&](size_t bytes) -> void* {
    void* r = (void*)p;
    p += (bytes + 255) & ~(size_t)255;
    return r;
  };
  const long NX = (long)B_ * S_ * D_;
  bf16_t* Xbf  = (bf16_t*)alloc(NX * 2);
  bf16_t* Wqt  = (bf16_t*)alloc((long)D_ * QD_ * 2);   // [QD_][D_]
  bf16_t* Wkt  = (bf16_t*)alloc((long)D_ * KD_ * 2);   // [KD_][D_]
  bf16_t* Wvt  = (bf16_t*)alloc((long)D_ * KD_ * 2);   // [KD_][D_]
  bf16_t* Wot  = (bf16_t*)alloc((long)QD_ * D_ * 2);   // [D_][QD_]
  float*  Qf   = (float*)alloc((long)B_ * S_ * QD_ * 4);
  float*  Kf   = (float*)alloc((long)B_ * S_ * KD_ * 4);
  float*  Vf   = (float*)alloc((long)B_ * S_ * KD_ * 4);
  bf16_t* Qbf  = (bf16_t*)alloc((long)B_ * S_ * QD_ * 2);
  bf16_t* Ksel = (bf16_t*)alloc((long)B_ * S_ * KD_ * 2);
  bf16_t* Vt   = (bf16_t*)alloc((long)B_ * S_ * KD_ * 2);
  int*    ordr = (int*)alloc((long)B_ * S_ * 4);
  int*    psel = (int*)alloc((long)B_ * S_ * 4);
  bf16_t* AOb  = (bf16_t*)alloc((long)B_ * S_ * QD_ * 2);

  // 1) activations fp32 -> bf16; weights fp32 -> bf16 transposed
  cvt_bf16_kernel<<<(int)(NX / 256), 256, 0, stream>>>(hidden, Xbf, (int)NX);
  dim3 tb(32, 8);
  cvt_transpose_kernel<<<dim3(D_ / 32, QD_ / 32), tb, 0, stream>>>(wq, Wqt, D_, QD_);
  cvt_transpose_kernel<<<dim3(D_ / 32, KD_ / 32), tb, 0, stream>>>(wk, Wkt, D_, KD_);
  cvt_transpose_kernel<<<dim3(D_ / 32, KD_ / 32), tb, 0, stream>>>(wv, Wvt, D_, KD_);
  cvt_transpose_kernel<<<dim3(QD_ / 32, D_ / 32), tb, 0, stream>>>(wo, Wot, QD_, D_);

  // 2) QKV projections: block tile = 128x64
  const int M = B_ * S_;
  gemm_nt_kernel<<<(M / 128) * (QD_ / 64), 256, 0, stream>>>(Xbf, Wqt, Qf, M, QD_, D_);
  gemm_nt_kernel<<<(M / 128) * (KD_ / 64), 256, 0, stream>>>(Xbf, Wkt, Kf, M, KD_, D_);
  gemm_nt_kernel<<<(M / 128) * (KD_ / 64), 256, 0, stream>>>(Xbf, Wvt, Vf, M, KD_, D_);

  // 3) active-token stable partition
  build_order_kernel<<<B_, 1024, 0, stream>>>(amask, pos_ids, ordr, psel);

  // 4) RoPE(Q); gather + RoPE(K); gather V transposed
  rope_q_kernel<<<(int)(((long)B_ * S_ * H_ * 64) / 256), 256, 0, stream>>>(Qf, cosT, sinT, Qbf);
  gather_kv_kernel<<<(int)(((long)B_ * S_ * HKV_ * 64) / 256), 256, 0, stream>>>(
      Kf, Vf, ordr, psel, cosT, sinT, Ksel, Vt);

  // 5) flash attention
  const float scale = 0.08838834764831845f;   // HD^-0.5
  attn_kernel<<<(B_ * H_ * (S_ / 16)) / ATT_WAVES, 128, 0, stream>>>(
      Qbf, Ksel, Vt, psel, scale, AOb);

  // 6) output projection -> d_out (fp32)
  gemm_nt_kernel<<<(M / 128) * (D_ / 64), 256, 0, stream>>>(AOb, Wot, out, M, D_, QD_);
}
